// GNN2_73040213836198
// MI455X (gfx1250) — compile-verified, matrix operand order checked by reading, not run on hardware
//
#include <hip/hip_runtime.h>

typedef __attribute__((ext_vector_type(16))) _Float16 v16h;
typedef __attribute__((ext_vector_type(8)))  float    v8f;

constexpr int NNODES = 30;
constexpr int NEDGES = NNODES - 1;   // 29 edges per destination (complete graph, no self)
constexpr int HID    = 32;

// Packed f16 weight blob layout inside d_ws (offsets in halves; all 16B aligned):
//   Wt1 [32][32]  @ 0      (layer1 Wa^T, K padded 6->32)
//   Wt2 [32][32]  @ 1024
//   Wt3 [32][32]  @ 2048
//   Wt4 [32][64]  @ 3072
//   Wt5 [32][32]  @ 5120
//   Wt6 [16][32]  @ 6144   (rows >= 3 zeroed)
// total 6656 halves = 13312 bytes, then 192 f32 biases, then the h buffer.
constexpr int OFF_WT1 = 0,    OFF_WT2 = 1024, OFF_WT3 = 2048;
constexpr int OFF_WT4 = 3072, OFF_WT5 = 5120, OFF_WT6 = 6144;
constexpr int WBLOB_HALVES = 6656;
constexpr size_t BIAS_BYTE_OFF = 13312;          // 832 * 16
constexpr size_t H_BYTE_OFF    = 14080;          // 13312 + 192*4, 64B aligned

// ---------------------------------------------------------------------------
// One-time weight pack: f32 -> f16, transpose to [n][k], zero-pad. Tiny kernel.
// ---------------------------------------------------------------------------
__global__ __launch_bounds__(256)
void pack_weights(const float* __restrict__ W1, const float* __restrict__ b1,
                  const float* __restrict__ W2, const float* __restrict__ b2,
                  const float* __restrict__ W3, const float* __restrict__ b3,
                  const float* __restrict__ W4, const float* __restrict__ b4,
                  const float* __restrict__ W5, const float* __restrict__ b5,
                  const float* __restrict__ W6, const float* __restrict__ b6,
                  _Float16* __restrict__ wblob, float* __restrict__ bblob) {
  const int t = threadIdx.x;
  for (int i = t; i < 32 * 32; i += 256) {       // Wt1: Wa (6,32) -> [n][32], pad
    const int n = i >> 5, k = i & 31;
    wblob[OFF_WT1 + n * 32 + k] = (_Float16)((k < 6) ? W1[k * HID + n] : 0.f);
  }
  for (int i = t; i < 32 * 32; i += 256) {       // Wt2
    const int n = i >> 5, k = i & 31;
    wblob[OFF_WT2 + n * 32 + k] = (_Float16)W2[k * HID + n];
  }
  for (int i = t; i < 32 * 32; i += 256) {       // Wt3
    const int n = i >> 5, k = i & 31;
    wblob[OFF_WT3 + n * 32 + k] = (_Float16)W3[k * HID + n];
  }
  for (int i = t; i < 32 * 64; i += 256) {       // Wt4: Wa2 (64,32) -> [n][64]
    const int n = i >> 6, k = i & 63;
    wblob[OFF_WT4 + n * 64 + k] = (_Float16)W4[k * HID + n];
  }
  for (int i = t; i < 32 * 32; i += 256) {       // Wt5
    const int n = i >> 5, k = i & 31;
    wblob[OFF_WT5 + n * 32 + k] = (_Float16)W5[k * HID + n];
  }
  for (int i = t; i < 16 * 32; i += 256) {       // Wt6: (32,3) -> [16][32], pad n
    const int n = i >> 5, k = i & 31;
    wblob[OFF_WT6 + n * 32 + k] = (_Float16)((n < 3) ? W6[k * 3 + n] : 0.f);
  }
  if (t < 32) {
    bblob[  0 + t] = b1[t];
    bblob[ 32 + t] = b2[t];
    bblob[ 64 + t] = b3[t];
    bblob[ 96 + t] = b4[t];
    bblob[128 + t] = b5[t];
    bblob[160 + t] = (t < 3) ? b6[t] : 0.f;
  }
}

// ---------------------------------------------------------------------------
// WMMA fragment gather from a row-major [rows][ld] f16 LDS tile.
// Per CDNA5 ISA 16-bit A-matrix layout (CDNA5 ISA §7.12.2):
//   VGPR i, lane l holds K = 16*(i>>2) + 8*(l>=16) + 2*(i&3) (+1), row = l&15.
// B uses the same pattern with N as the row, so W is stored transposed.
// ---------------------------------------------------------------------------
__device__ __forceinline__ v16h load_frag(const _Float16* base, int ld) {
  const int lane = threadIdx.x & 31;
  const int row  = lane & 15;
  const int kb   = (lane & 16) ? 8 : 0;
  union { v16h v; unsigned u[8]; } f;
  const unsigned* rp = (const unsigned*)(base + row * ld);
  #pragma unroll
  for (int i = 0; i < 8; ++i) {
    const int k = ((i >> 2) << 4) + kb + ((i & 3) << 1);
    f.u[i] = rp[k >> 1];
  }
  return f.v;
}

// H[32][NOUT] = relu(A[32][K] @ Wt[NOUT][K]^T + bias); f32 accum, f16 to LDS.
template<int K, int NOUT>
__device__ __forceinline__ void gemm_relu_store(const _Float16* A, const _Float16* Wt,
                                                const float* bias, _Float16* H) {
  const int lane = threadIdx.x & 31;
  const int hi8  = (lane & 16) ? 8 : 0;
  #pragma unroll
  for (int nt = 0; nt < NOUT / 16; ++nt) {
    #pragma unroll
    for (int mt = 0; mt < 2; ++mt) {
      v8f acc = {};
      #pragma unroll
      for (int kt = 0; kt < K / 32; ++kt) {
        v16h a = load_frag(A  + mt * 16 * K + kt * 32, K);
        v16h w = load_frag(Wt + nt * 16 * K + kt * 32, K);
        acc = __builtin_amdgcn_wmma_f32_16x16x32_f16(false, a, false, w,
                                                     (short)0, acc, false, false);
      }
      const int   n  = (lane & 15) + nt * 16;
      const float bv = bias[n];
      #pragma unroll
      for (int r = 0; r < 8; ++r) {
        float v = acc[r] + bv;
        v = v > 0.f ? v : 0.f;
        const int m = mt * 16 + hi8 + r;                 // D-tile row -> edge row
        H[m * NOUT + n] = (_Float16)v;
      }
    }
  }
}

// Final linear fused with mean aggregation: D rows are edges; sum accumulator
// registers over valid rows (m < 29), fold lane halves via xor-16 shuffle.
template<int NT, int OUTD>
__device__ __forceinline__ void gemm_reduce_out(const _Float16* A, const _Float16* Wt,
                                                const float* bias, float* out, float scale) {
  const int lane = threadIdx.x & 31;
  const int hi8  = (lane & 16) ? 8 : 0;
  #pragma unroll
  for (int nt = 0; nt < NT; ++nt) {
    const int   n  = (lane & 15) + nt * 16;
    const float bv = (n < OUTD) ? bias[n] : 0.f;
    float colsum = 0.f;
    #pragma unroll
    for (int mt = 0; mt < 2; ++mt) {
      v8f acc = {};
      v16h a = load_frag(A  + mt * 16 * 32, 32);
      v16h w = load_frag(Wt + nt * 16 * 32, 32);
      acc = __builtin_amdgcn_wmma_f32_16x16x32_f16(false, a, false, w,
                                                   (short)0, acc, false, false);
      #pragma unroll
      for (int r = 0; r < 8; ++r) {
        const int m = mt * 16 + hi8 + r;
        if (m < NEDGES) colsum += acc[r] + bv;           // mask padded edge rows
      }
    }
    colsum += __shfl_xor(colsum, 16, 32);                // combine lane halves
    if (lane < 16 && n < OUTD) out[n] = colsum * scale;
  }
}

// ---------------------------------------------------------------------------
// Fused EdgeConv. 4 waves/block share one LDS copy of the (pre-packed f16)
// weights; each wave independently owns one (batch, dst) task.
// ---------------------------------------------------------------------------
template<int NODE_F, int K1, int OUTD, int NT3>
__global__ __launch_bounds__(128)
void edgeconv_kernel(const float* __restrict__ feat,
                     const _Float16* __restrict__ WtA, const _Float16* __restrict__ WtB,
                     const _Float16* __restrict__ WtC,
                     const float* __restrict__ bA, const float* __restrict__ bB,
                     const float* __restrict__ bC,
                     float* __restrict__ out, int ntask) {
  __shared__ __align__(16) _Float16 sWtA[32 * K1];
  __shared__ __align__(16) _Float16 sWtB[32 * 32];
  __shared__ __align__(16) _Float16 sWtC[NT3 * 16 * 32];
  __shared__ float sbA[32], sbB[32], sbC[32];
  __shared__ __align__(16) _Float16 Abuf[4][32 * K1];    // per-wave edge rows
  __shared__ __align__(16) _Float16 H1[4][32 * 32];
  __shared__ __align__(16) _Float16 H2[4][32 * 32];

  const int t    = threadIdx.x;
  const int wv   = t >> 5;                               // wave id 0..3
  const int lane = t & 31;

  // Cooperative weight staging: pure 16B f16 copies (no cvt, no transpose).
  {
    const uint4* s;
    uint4* d;
    s = (const uint4*)WtA; d = (uint4*)sWtA;
    for (int i = t; i < 32 * K1 / 8; i += 128) d[i] = s[i];
    s = (const uint4*)WtB; d = (uint4*)sWtB;
    for (int i = t; i < 32 * 32 / 8; i += 128) d[i] = s[i];
    s = (const uint4*)WtC; d = (uint4*)sWtC;
    for (int i = t; i < NT3 * 16 * 32 / 8; i += 128) d[i] = s[i];
    if (t < 32) { sbA[t] = bA[t]; sbB[t] = bB[t]; sbC[t] = bC[t]; }
  }

  // Per-wave task (uniform per wave; clamp keeps waves convergent for WMMA).
  int bidx = blockIdx.x * 4 + wv;
  if (bidx >= ntask) bidx = ntask - 1;
  const int b   = bidx / NNODES;
  const int dst = bidx % NNODES;

  // Build the 29 edge rows [xi, xj - xi] for this dst (rows 29..31 zeroed).
  _Float16* A = Abuf[wv];
  {
    const float* xb = feat + (size_t)b * NNODES * NODE_F;
    const float* xi = xb + dst * NODE_F;
    const int e = lane;
    if (e < NEDGES) {
      const int s = (e < dst) ? e : e + 1;               // all src != dst
      const float* xj = xb + s * NODE_F;
      #pragma unroll
      for (int k = 0; k < NODE_F; ++k) {
        const float a = xi[k];
        A[e * K1 + k]          = (_Float16)a;            // xi
        A[e * K1 + NODE_F + k] = (_Float16)(xj[k] - a);  // xj - xi
      }
      for (int k = 2 * NODE_F; k < K1; ++k) A[e * K1 + k] = (_Float16)0.f;
    } else {
      for (int k = 0; k < K1; ++k) A[e * K1 + k] = (_Float16)0.f;
    }
  }
  __syncthreads();   // weights + own A visible; afterwards each wave touches
                     // only its private buffers (same-wave LDS is in-order).

  gemm_relu_store<K1, 32>(A,      sWtA, sbA, H1[wv]);    // linear1 + relu
  gemm_relu_store<32, 32>(H1[wv], sWtB, sbB, H2[wv]);    // linear2 + relu
  gemm_reduce_out<NT3, OUTD>(H2[wv], sWtC, sbC,          // linear3 + mean-agg
                             out + (size_t)bidx * OUTD, 1.0f / NEDGES);
}

extern "C" void kernel_launch(void* const* d_in, const int* in_sizes, int n_in,
                              void* d_out, int out_size, void* d_ws, size_t ws_size,
                              hipStream_t stream) {
  const float* x  = (const float*)d_in[0];
  const float* W1 = (const float*)d_in[1];
  const float* b1 = (const float*)d_in[2];
  const float* W2 = (const float*)d_in[3];
  const float* b2 = (const float*)d_in[4];
  const float* W3 = (const float*)d_in[5];
  const float* b3 = (const float*)d_in[6];
  const float* W4 = (const float*)d_in[7];
  const float* b4 = (const float*)d_in[8];
  const float* W5 = (const float*)d_in[9];
  const float* b5 = (const float*)d_in[10];
  const float* W6 = (const float*)d_in[11];
  const float* b6 = (const float*)d_in[12];
  // edge_index (d_in[13]) unused: complete graph structure is compile-time known.

  const int B     = in_sizes[0] / (NNODES * 3);          // 1024
  const int ntask = B * NNODES;                          // 30720

  _Float16* wblob = (_Float16*)d_ws;
  float*    bblob = (float*)((char*)d_ws + BIAS_BYTE_OFF);
  float*    h     = (float*)((char*)d_ws + H_BYTE_OFF);  // [B][N][32] fp32
  float*    out   = (float*)d_out;                       // [B][N][3]  fp32

  pack_weights<<<1, 256, 0, stream>>>(W1, b1, W2, b2, W3, b3,
                                      W4, b4, W5, b5, W6, b6, wblob, bblob);

  dim3 blk(128);
  dim3 grid((ntask + 3) / 4);
  // Layer 1: node feat 3, concat 6 -> K pad 32, out 32 (2 N-tiles in final GEMM)
  edgeconv_kernel<3, 32, 32, 2><<<grid, blk, 0, stream>>>(
      x, wblob + OFF_WT1, wblob + OFF_WT2, wblob + OFF_WT3,
      bblob + 0, bblob + 32, bblob + 64, h, ntask);
  // Layer 2: node feat 32, concat 64, out 3 (1 N-tile in final GEMM)
  edgeconv_kernel<32, 64, 3, 1><<<grid, blk, 0, stream>>>(
      h, wblob + OFF_WT4, wblob + OFF_WT5, wblob + OFF_WT6,
      bblob + 96, bblob + 128, bblob + 160, out, ntask);
}
